// EulerCEAttention_40991167873348
// MI455X (gfx1250) — matrix operand order repaired and checked by reading, not compiled, not run on hardware
//
#include <hip/hip_runtime.h>
#include <hip/hip_bf16.h>
#include <math.h>

// ---------- types ----------
typedef __attribute__((ext_vector_type(16))) __bf16 v16bf;
typedef __attribute__((ext_vector_type(8)))  __bf16 v8bf;
typedef __attribute__((ext_vector_type(8)))  float  v8f;

#define B_SZ   2
#define SEQ_N  2048
#define DMODEL 1024
#define NHEAD  16
#define DHEAD  64
#define MROWS  (B_SZ * SEQ_N)          // 4096

// ---------- CDNA5 async global->LDS copy (ASYNCcnt-tracked) ----------
// Per ISA 10.2 aperture rules, the low 32 bits of a generic LDS pointer are
// the LDS byte offset, which is what VDST of the async op wants.
__device__ __forceinline__ unsigned lds_off_u32(const void* p) {
    return (unsigned)(size_t)p;
}
__device__ __forceinline__ void async_ld_b128(unsigned lds, const void* g) {
    asm volatile("global_load_async_to_lds_b128 %0, %1, off"
                 :: "v"(lds), "v"(g) : "memory");
}
__device__ __forceinline__ void wait_async0() {
    asm volatile("s_wait_asynccnt 0x0" ::: "memory");
}

// ---------- DPP 16-lane row all-reduce (no LDS, pure VALU) ----------
template <int CTRL>
__device__ __forceinline__ float dpp16(float v) {
    int r = __builtin_amdgcn_update_dpp(0, __builtin_bit_cast(int, v),
                                        CTRL, 0xf, 0xf, true);
    return __builtin_bit_cast(float, r);
}
__device__ __forceinline__ float row16_max(float t) {
    t = fmaxf(t, dpp16<0x121>(t));
    t = fmaxf(t, dpp16<0x122>(t));
    t = fmaxf(t, dpp16<0x124>(t));
    t = fmaxf(t, dpp16<0x128>(t));
    return t;
}
__device__ __forceinline__ float row16_sum(float t) {
    t += dpp16<0x121>(t);
    t += dpp16<0x122>(t);
    t += dpp16<0x124>(t);
    t += dpp16<0x128>(t);
    return t;
}

// =====================================================================
// fp32 -> bf16 conversion
// =====================================================================
__global__ void f32_to_bf16_kernel(const float* __restrict__ in,
                                   __bf16* __restrict__ out, int n) {
    int i = blockIdx.x * blockDim.x + threadIdx.x;
    if (i < n) out[i] = (__bf16)in[i];
}

// =====================================================================
// WMMA GEMM:  C[m][n] = sum_k A[m*K+k] * B[n*K+k]   (C = A * B^T)
// block tile 128x64, 8 waves (4x2), 32x32 per wave, BK = 32.
// Double-buffered LDS fed by async global->LDS copies.
// =====================================================================
template <bool OUT_BF16>
__global__ __launch_bounds__(256)
void wmma_gemm_nt(const __bf16* __restrict__ A, const __bf16* __restrict__ B,
                  void* __restrict__ Cout, int M, int N, int K) {
    constexpr int BM = 128, BN = 64, BK = 32;
    constexpr int LDA = BK + 8;   // 40 elements, 80B
    constexpr int LDB = BK + 8;
    __shared__ __align__(16) __bf16 sA[2][BM * LDA];
    __shared__ __align__(16) __bf16 sB[2][BN * LDB];

    const int tid  = threadIdx.x;
    const int wave = tid >> 5;
    const int lane = tid & 31;
    const int l16  = lane & 15;
    const int lh   = lane >> 4;
    const int wm   = wave & 3;
    const int wn   = wave >> 2;
    const int bm   = blockIdx.x * BM;
    const int bn   = blockIdx.y * BN;

    // per-thread staging coordinates (3 x 16B async copies per K-slab)
    const int ra0 = (tid * 2) >> 2,     ca0 = ((tid * 2) & 3) << 3;
    const int ra1 = (tid * 2 + 1) >> 2, ca1 = ((tid * 2 + 1) & 3) << 3;
    const int rb  = tid >> 2,           cb  = (tid & 3) << 3;

    auto stage = [&](int buf, int k0) {
        async_ld_b128(lds_off_u32(&sA[buf][ra0 * LDA + ca0]),
                      A + (size_t)(bm + ra0) * K + k0 + ca0);
        async_ld_b128(lds_off_u32(&sA[buf][ra1 * LDA + ca1]),
                      A + (size_t)(bm + ra1) * K + k0 + ca1);
        async_ld_b128(lds_off_u32(&sB[buf][rb * LDB + cb]),
                      B + (size_t)(bn + rb) * K + k0 + cb);
    };

    v8f acc[2][2];
#pragma unroll
    for (int i = 0; i < 2; ++i)
#pragma unroll
        for (int j = 0; j < 2; ++j)
#pragma unroll
            for (int g = 0; g < 8; ++g) acc[i][j][g] = 0.0f;

    stage(0, 0);
    wait_async0();
    __syncthreads();

    int buf = 0;
    for (int k0 = 0; k0 < K; k0 += BK) {
        if (k0 + BK < K) stage(buf ^ 1, k0 + BK);   // overlap copy w/ compute

        v16bf afr[2], bfr[2];
#pragma unroll
        for (int i = 0; i < 2; ++i) {
            int row = wm * 32 + i * 16 + l16;
            int kc  = lh * 8;
            v8bf lo = *(v8bf*)&sA[buf][row * LDA + kc];
            v8bf hi = *(v8bf*)&sA[buf][row * LDA + kc + 16];
#pragma unroll
            for (int e = 0; e < 8; ++e) { afr[i][e] = lo[e]; afr[i][8 + e] = hi[e]; }
        }
#pragma unroll
        for (int j = 0; j < 2; ++j) {
            int col = wn * 32 + j * 16 + l16;
            int kc  = lh * 8;
            v8bf lo = *(v8bf*)&sB[buf][col * LDB + kc];
            v8bf hi = *(v8bf*)&sB[buf][col * LDB + kc + 16];
#pragma unroll
            for (int e = 0; e < 8; ++e) { bfr[j][e] = lo[e]; bfr[j][8 + e] = hi[e]; }
        }
#pragma unroll
        for (int i = 0; i < 2; ++i)
#pragma unroll
            for (int j = 0; j < 2; ++j)
                acc[i][j] = __builtin_amdgcn_wmma_f32_16x16x32_bf16(
                    false, afr[i], false, bfr[j], (short)0, acc[i][j], false, false);

        wait_async0();
        __syncthreads();
        buf ^= 1;
    }

    // epilogue: C layout -> lane=col, VGPR g = row (g + 8*lh)
#pragma unroll
    for (int i = 0; i < 2; ++i)
#pragma unroll
        for (int j = 0; j < 2; ++j) {
            int col = bn + wn * 32 + j * 16 + l16;
#pragma unroll
            for (int g = 0; g < 8; ++g) {
                int row = bm + wm * 32 + i * 16 + g + lh * 8;
                float v = acc[i][j][g];
                if (OUT_BF16)
                    ((__bf16*)Cout)[(size_t)row * N + col] = (__bf16)v;
                else
                    ((float*)Cout)[(size_t)row * N + col] = v;
            }
        }
}

// =====================================================================
// RoPE(pi-frequency, gated) + pack:
//   qkv (B,N,3,H,dh) bf16  ->  Q,K (B,H,N,dh) bf16,  Vt (B,H,dh,N) bf16
// =====================================================================
__global__ __launch_bounds__(256)
void rope_pack_kernel(const __bf16* __restrict__ qkv,
                      const float* __restrict__ pi_gate_logit,
                      __bf16* __restrict__ Q, __bf16* __restrict__ Kd,
                      __bf16* __restrict__ Vt) {
    int idx = blockIdx.x * blockDim.x + threadIdx.x;   // (b,n,h,p)
    int p = idx & 31;
    int h = (idx >> 5) & (NHEAD - 1);
    int n = (idx >> 9) & (SEQ_N - 1);
    int b = idx >> 20;
    if (b >= B_SZ) return;

    float pi_g = 1.0f / (1.0f + __expf(-pi_gate_logit[h]));
    float kfrac = (float)p * (1.0f / 32.0f);
    float inv_freq = __expf((1.0f - kfrac) * 1.14472988584940017f); // pi^(1-kfrac)
    float theta = (float)n * inv_freq * pi_g;
    float s, c;
    __sincosf(theta, &s, &c);

    size_t base = ((size_t)(b * SEQ_N + n)) * (3 * DMODEL) + h * DHEAD + 2 * p;
    float q1 = (float)qkv[base];
    float q2 = (float)qkv[base + 1];
    float k1 = (float)qkv[base + DMODEL];
    float k2 = (float)qkv[base + DMODEL + 1];
    float v1 = (float)qkv[base + 2 * DMODEL];
    float v2 = (float)qkv[base + 2 * DMODEL + 1];

    size_t qk = ((size_t)((b * NHEAD + h)) * SEQ_N + n) * DHEAD + 2 * p;
    Q[qk]     = (__bf16)(q1 * c - q2 * s);
    Q[qk + 1] = (__bf16)(q1 * s + q2 * c);
    Kd[qk]     = (__bf16)(k1 * c - k2 * s);
    Kd[qk + 1] = (__bf16)(k1 * s + k2 * c);

    size_t vt = ((size_t)((b * NHEAD + h) * DHEAD + 2 * p)) * SEQ_N + n;
    Vt[vt]         = (__bf16)v1;
    Vt[vt + SEQ_N] = (__bf16)v2;
}

// =====================================================================
// Flash attention (causal + distance-decay bias), WMMA bf16, base-2
// online softmax, DPP row reductions, async double-buffered K/V tiles.
// 128 threads (4 waves); each wave owns 16 query rows; 32-key tiles.
// =====================================================================
__global__ __launch_bounds__(128)
void flash_attn_kernel(const __bf16* __restrict__ Q, const __bf16* __restrict__ Kg,
                       const __bf16* __restrict__ Vt,
                       const float* __restrict__ log_xi,
                       const float* __restrict__ e_gate_logit,
                       __bf16* __restrict__ O) {
    constexpr int QT = 64, KT = 32;
    constexpr int LDK = DHEAD + 8;   // 72
    constexpr int LDV = KT + 8;      // 40
    __shared__ __align__(16) __bf16 sK[2][KT * LDK];     // K tile  32 x 64
    __shared__ __align__(16) __bf16 sV[2][DHEAD * LDV];  // Vt tile 64 x 32
    __shared__ __align__(16) float  sP[4][16 * 32];      // per-wave P scratch

    const int tid  = threadIdx.x;
    const int wave = tid >> 5;
    const int lane = tid & 31;
    const int l16  = lane & 15;
    const int lh   = lane >> 4;

    const int nqt = SEQ_N / QT;                // 32
    int bid = blockIdx.x;
    int qt  = bid % nqt;
    int h   = (bid / nqt) % NHEAD;
    int b   = bid / (nqt * NHEAD);

    const float LOG2E = 1.4426950408889634f;
    float xi     = __expf(log_xi[h]);
    float e_g    = 1.0f / (1.0f + __expf(-e_gate_logit[h]));
    float slope2 = (e_g / xi) * LOG2E;          // decay slope in log2 space
    float scale2 = 0.125f * LOG2E;              // 1/sqrt(64) * log2(e)

    const int qbase = qt * QT + wave * 16;

    const __bf16* Kbase  = Kg + (size_t)(b * NHEAD + h) * SEQ_N * DHEAD;
    const __bf16* Vtbase = Vt + (size_t)(b * NHEAD + h) * DHEAD * SEQ_N;

    // per-thread staging coordinates (2 + 2 x 16B async copies per tile)
    const int rk0 = (tid * 2) >> 3,     ck0 = ((tid * 2) & 7) << 3;
    const int rk1 = (tid * 2 + 1) >> 3, ck1 = ((tid * 2 + 1) & 7) << 3;
    const int rv0 = (tid * 2) >> 2,     cv0 = ((tid * 2) & 3) << 3;
    const int rv1 = (tid * 2 + 1) >> 2, cv1 = ((tid * 2 + 1) & 3) << 3;

    auto stageKV = [&](int buf, int kt) {
        async_ld_b128(lds_off_u32(&sK[buf][rk0 * LDK + ck0]),
                      Kbase + (size_t)(kt * KT + rk0) * DHEAD + ck0);
        async_ld_b128(lds_off_u32(&sK[buf][rk1 * LDK + ck1]),
                      Kbase + (size_t)(kt * KT + rk1) * DHEAD + ck1);
        async_ld_b128(lds_off_u32(&sV[buf][rv0 * LDV + cv0]),
                      Vtbase + (size_t)rv0 * SEQ_N + kt * KT + cv0);
        async_ld_b128(lds_off_u32(&sV[buf][rv1 * LDV + cv1]),
                      Vtbase + (size_t)rv1 * SEQ_N + kt * KT + cv1);
    };

    // Q fragments (A layout), held for all key tiles
    const __bf16* qp = Q + ((size_t)(b * NHEAD + h) * SEQ_N + qbase + l16) * DHEAD;
    v16bf qfr[2];
#pragma unroll
    for (int s2 = 0; s2 < 2; ++s2) {
        int kc = s2 * 32 + lh * 8;
        v8bf lo = *(const v8bf*)(qp + kc);
        v8bf hi = *(const v8bf*)(qp + kc + 16);
#pragma unroll
        for (int e = 0; e < 8; ++e) { qfr[s2][e] = lo[e]; qfr[s2][8 + e] = hi[e]; }
    }

    v8f ofr[4];
    float mrow[8], lrow[8], qsf[8];
#pragma unroll
    for (int t = 0; t < 4; ++t)
#pragma unroll
        for (int g = 0; g < 8; ++g) ofr[t][g] = 0.0f;
#pragma unroll
    for (int g = 0; g < 8; ++g) {
        mrow[g] = -3.0e38f;
        lrow[g] = 0.0f;
        qsf[g]  = (float)(qbase + g + lh * 8);
    }

    const int ktmax = (qt * QT + QT) / KT;     // causal bound
    stageKV(0, 0);
    wait_async0();
    __syncthreads();

    int buf = 0;
    for (int kt = 0; kt < ktmax; ++kt) {
        if (kt + 1 < ktmax) stageKV(buf ^ 1, kt + 1);   // overlap copy w/ compute

        // S = Q * K^T  (two 16-col tiles, two k-steps over dh)
        v8f stile[2];
#pragma unroll
        for (int j = 0; j < 2; ++j) {
#pragma unroll
            for (int g = 0; g < 8; ++g) stile[j][g] = 0.0f;
#pragma unroll
            for (int s2 = 0; s2 < 2; ++s2) {
                int key = j * 16 + l16;
                int kc  = s2 * 32 + lh * 8;
                v8bf lo = *(v8bf*)&sK[buf][key * LDK + kc];
                v8bf hi = *(v8bf*)&sK[buf][key * LDK + kc + 16];
                v16bf kfr;
#pragma unroll
                for (int e = 0; e < 8; ++e) { kfr[e] = lo[e]; kfr[8 + e] = hi[e]; }
                stile[j] = __builtin_amdgcn_wmma_f32_16x16x32_bf16(
                    false, qfr[s2], false, kfr, (short)0, stile[j], false, false);
            }
            // scale + decay bias + causal mask (all in log2 space)
            float kfv = (float)(kt * KT + j * 16 + l16);
#pragma unroll
            for (int g = 0; g < 8; ++g) {
                float val = stile[j][g] * scale2 - (qsf[g] - kfv) * slope2;
                stile[j][g] = (kfv <= qsf[g]) ? val : -3.0e38f;
            }
        }

        // online softmax (DPP 16-lane row reductions)
        float alpha[8];
#pragma unroll
        for (int g = 0; g < 8; ++g) {
            float t  = row16_max(fmaxf(stile[0][g], stile[1][g]));
            float nm = fmaxf(mrow[g], t);
            alpha[g] = __builtin_amdgcn_exp2f(mrow[g] - nm);
            mrow[g]  = nm;
        }
#pragma unroll
        for (int j = 0; j < 2; ++j)
#pragma unroll
            for (int g = 0; g < 8; ++g)
                stile[j][g] = __builtin_amdgcn_exp2f(stile[j][g] - mrow[g]);
#pragma unroll
        for (int g = 0; g < 8; ++g) {
            float rs = row16_sum(stile[0][g] + stile[1][g]);
            lrow[g] = lrow[g] * alpha[g] + rs;
        }
#pragma unroll
        for (int t = 0; t < 4; ++t)
#pragma unroll
            for (int g = 0; g < 8; ++g) ofr[t][g] *= alpha[g];

        // P: C-layout -> A-layout transpose through per-wave LDS scratch
        float* pw = sP[wave];
#pragma unroll
        for (int j = 0; j < 2; ++j)
#pragma unroll
            for (int g = 0; g < 8; ++g)
                pw[(g + lh * 8) * 32 + j * 16 + l16] = stile[j][g];
        asm volatile("s_wait_dscnt 0" ::: "memory");

        v16bf pfr;
#pragma unroll
        for (int e = 0; e < 8; ++e) {
            pfr[e]     = (__bf16)pw[l16 * 32 + lh * 8 + e];
            pfr[8 + e] = (__bf16)pw[l16 * 32 + lh * 8 + 16 + e];
        }

        // O += P * V  (4 dh tiles; B-frag rows of Vt are contiguous in key)
#pragma unroll
        for (int t = 0; t < 4; ++t) {
            int dcol = t * 16 + l16;
            int kc   = lh * 8;
            v8bf lo = *(v8bf*)&sV[buf][dcol * LDV + kc];
            v8bf hi = *(v8bf*)&sV[buf][dcol * LDV + kc + 16];
            v16bf vfr;
#pragma unroll
            for (int e = 0; e < 8; ++e) { vfr[e] = lo[e]; vfr[8 + e] = hi[e]; }
            ofr[t] = __builtin_amdgcn_wmma_f32_16x16x32_bf16(
                false, pfr, false, vfr, (short)0, ofr[t], false, false);
        }

        wait_async0();
        __syncthreads();
        buf ^= 1;
    }

    // normalize and store O as bf16 into (B,N,H*dh) for the output GEMM
#pragma unroll
    for (int t = 0; t < 4; ++t) {
        int dcol = t * 16 + l16;
#pragma unroll
        for (int g = 0; g < 8; ++g) {
            int qidx  = qbase + g + lh * 8;
            float val = ofr[t][g] / lrow[g];
            O[((size_t)(b * SEQ_N + qidx)) * DMODEL + h * DHEAD + dcol] = (__bf16)val;
        }
    }
}

// =====================================================================
// host-side launcher
// =====================================================================
extern "C" void kernel_launch(void* const* d_in, const int* in_sizes, int n_in,
                              void* d_out, int out_size, void* d_ws, size_t ws_size,
                              hipStream_t stream) {
    const float* x        = (const float*)d_in[0];
    const float* Wqkv     = (const float*)d_in[1];
    const float* Wo       = (const float*)d_in[2];
    const float* log_xi   = (const float*)d_in[3];
    const float* pi_gate  = (const float*)d_in[4];
    const float* e_gate   = (const float*)d_in[5];

    char* ws = (char*)d_ws;
    size_t off = 0;
    auto carve = [&](size_t bytes) -> char* {
        char* p = ws + off;
        off += (bytes + 255) & ~(size_t)255;
        return p;
    };

    const int nX    = MROWS * DMODEL;          // 4,194,304
    const int nWqkv = 3 * DMODEL * DMODEL;     // 3,145,728
    const int nWo   = DMODEL * DMODEL;         // 1,048,576

    __bf16* xb    = (__bf16*)carve((size_t)nX * 2);
    __bf16* wqkvb = (__bf16*)carve((size_t)nWqkv * 2);
    __bf16* wob   = (__bf16*)carve((size_t)nWo * 2);
    __bf16* qkvb  = (__bf16*)carve((size_t)MROWS * 3 * DMODEL * 2);
    __bf16* Qb    = (__bf16*)carve((size_t)nX * 2);
    __bf16* Kb    = (__bf16*)carve((size_t)nX * 2);
    __bf16* Vtb   = (__bf16*)carve((size_t)nX * 2);
    __bf16* Ob    = (__bf16*)carve((size_t)nX * 2);

    // 1) convert inputs to bf16
    f32_to_bf16_kernel<<<(nX + 255) / 256, 256, 0, stream>>>(x, xb, nX);
    f32_to_bf16_kernel<<<(nWqkv + 255) / 256, 256, 0, stream>>>(Wqkv, wqkvb, nWqkv);
    f32_to_bf16_kernel<<<(nWo + 255) / 256, 256, 0, stream>>>(Wo, wob, nWo);

    // 2) qkv = x @ Wqkv^T   (M=4096, N=3072, K=1024) -> bf16
    {
        dim3 grid(MROWS / 128, (3 * DMODEL) / 64);
        wmma_gemm_nt<true><<<grid, 256, 0, stream>>>(xb, wqkvb, qkvb,
                                                     MROWS, 3 * DMODEL, DMODEL);
    }

    // 3) RoPE + pack (Q,K head-major; V transposed)
    {
        int total = B_SZ * SEQ_N * NHEAD * (DHEAD / 2);   // 2,097,152
        rope_pack_kernel<<<total / 256, 256, 0, stream>>>(qkvb, pi_gate, Qb, Kb, Vtb);
    }

    // 4) flash attention
    {
        int blocks = B_SZ * NHEAD * (SEQ_N / 64);         // 1024
        flash_attn_kernel<<<blocks, 128, 0, stream>>>(Qb, Kb, Vtb, log_xi, e_gate, Ob);
    }

    // 5) out = O @ Wo^T  (M=4096, N=1024, K=1024) -> fp32 d_out
    {
        dim3 grid(MROWS / 128, DMODEL / 64);
        wmma_gemm_nt<false><<<grid, 256, 0, stream>>>(Ob, wob, (float*)d_out,
                                                      MROWS, DMODEL, DMODEL);
    }
    (void)in_sizes; (void)n_in; (void)out_size; (void)ws_size;
}
